// RealVQVAE_34007551049844
// MI455X (gfx1250) — matrix operand order repaired and checked by reading, not compiled, not run on hardware
//
#include <hip/hip_runtime.h>

typedef __attribute__((ext_vector_type(16))) __bf16 v16bf;
typedef __attribute__((ext_vector_type(8)))  float  v8f;

#define WMMA_BF16(A, B, C) \
  __builtin_amdgcn_wmma_f32_16x16x32_bf16(false, (A), false, (B), (short)0, (C), false, false)

// Async global->LDS (CDNA5): bypasses VGPRs, tracked by ASYNCcnt.
#if defined(__has_builtin)
#if __has_builtin(__builtin_amdgcn_global_load_async_to_lds_b32) && \
    __has_builtin(__builtin_amdgcn_s_wait_asynccnt)
#define HAVE_ASYNC_LDS 1
#endif
#endif
#ifndef HAVE_ASYNC_LDS
#define HAVE_ASYNC_LDS 0
#endif

#define GLOBAL_AS __attribute__((address_space(1)))
#define LDS_AS    __attribute__((address_space(3)))

__device__ __forceinline__ void bf_split(float v, __bf16& hi, __bf16& lo) {
  __bf16 h = (__bf16)v;          // RTNE f32 -> bf16
  hi = h;
  lo = (__bf16)(v - (float)h);   // residual
}

// ---------------------------------------------------------------------------
// Pack conv weights [COUT][CIN][3] (f32) -> bf16 hi/lo, layout [co][kk*CIN+ci]
// so every 32-wide K block is a single kernel tap over contiguous channels.
// ---------------------------------------------------------------------------
__global__ void pack_w_kernel(const float* __restrict__ w,
                              __bf16* __restrict__ hi, __bf16* __restrict__ lo,
                              int COUT, int CIN) {
  int idx = blockIdx.x * 256 + threadIdx.x;
  int total = COUT * CIN * 3;
  if (idx >= total) return;
  int co = idx / (CIN * 3);
  int rem = idx % (CIN * 3);
  int ci = rem / 3;
  int kk = rem % 3;
  float v = w[idx];
  __bf16 h, l;
  bf_split(v, h, l);
  size_t dst = (size_t)co * (3 * CIN) + kk * CIN + ci;
  hi[dst] = h;
  lo[dst] = l;
}

// ---------------------------------------------------------------------------
// Pack codebook [4096][128] -> bf16 hi/lo and per-row squared norms.
// ---------------------------------------------------------------------------
__global__ void pack_emb_kernel(const float* __restrict__ emb,
                                __bf16* __restrict__ hi, __bf16* __restrict__ lo,
                                float* __restrict__ norms) {
  int m = blockIdx.x;
  int k = threadIdx.x;  // 128 threads
  float v = emb[(size_t)m * 128 + k];
  __bf16 h, l;
  bf_split(v, h, l);
  hi[(size_t)m * 128 + k] = h;
  lo[(size_t)m * 128 + k] = l;
  __shared__ float red[128];
  red[k] = v * v;
  __syncthreads();
  for (int s = 64; s > 0; s >>= 1) {
    if (k < s) red[k] += red[k + s];
    __syncthreads();
  }
  if (k == 0) norms[m] = red[0];
}

// ---------------------------------------------------------------------------
// Implicit-GEMM conv1d (pad 1) with bf16x3 WMMA, f32 accumulate.
// Block: 128 threads (4 waves). Tile: 64 Cout (16 per wave) x 16 positions.
// Input patch staged to LDS; interior tiles use the CDNA5 async direct-to-LDS
// path (no VGPR round-trip), boundary tiles use predicated loads for zero pad.
// MODE 0: +bias, BN(eval), LeakyReLU(0.01), store NCL.
// MODE 1: +bias only, store transposed [b*LOUT + l][128] (z layout for VQ).
// ---------------------------------------------------------------------------
template <int CIN, int COUT, int STRIDE, int MODE>
__global__ __launch_bounds__(128)
void conv_wmma_kernel(const float* __restrict__ xin,        // [B, CIN, LIN]
                      const __bf16* __restrict__ w_hi,
                      const __bf16* __restrict__ w_lo,      // [COUT][3*CIN]
                      const float* __restrict__ bias,
                      const float* __restrict__ bn_g, const float* __restrict__ bn_b,
                      const float* __restrict__ bn_m, const float* __restrict__ bn_v,
                      float* __restrict__ out,
                      int LIN, int LOUT) {
  constexpr int P = STRIDE * 15 + 3;  // input positions covered by a 16-wide tile
  __shared__ float xt[32][P + 1];

  const int tid  = threadIdx.x;
  const int lane = tid & 31;
  const int wv   = tid >> 5;
  const int half = lane >> 4;
  const int nr   = lane & 15;

  const int tilesPerImg = LOUT / 16;
  const int b  = blockIdx.x / tilesPerImg;
  const int l0 = (blockIdx.x % tilesPerImg) * 16;
  const int cobase = blockIdx.y * 64 + wv * 16;
  const int x0 = STRIDE * l0 - 1;  // tile origin in the input row

#if HAVE_ASYNC_LDS
  const bool interior = (x0 >= 0) && (x0 + P - 1 < LIN);
#endif

  v8f acc = {0.f, 0.f, 0.f, 0.f, 0.f, 0.f, 0.f, 0.f};

  for (int ci0 = 0; ci0 < CIN; ci0 += 32) {
    __syncthreads();
#if HAVE_ASYNC_LDS
    if (interior) {
      // async copy: global -> LDS directly, no VGPR data movement
      for (int idx = tid; idx < 32 * P; idx += 128) {
        int cr = idx / P;
        int p  = idx % P;
        float* gsrc = const_cast<float*>(xin) +
                      ((size_t)b * CIN + ci0 + cr) * LIN + x0 + p;
        __builtin_amdgcn_global_load_async_to_lds_b32(
            (GLOBAL_AS int*)gsrc, (LDS_AS int*)&xt[cr][p], 0, 0);
      }
      __builtin_amdgcn_s_wait_asynccnt(0);
    } else
#endif
    {
      for (int idx = tid; idx < 32 * P; idx += 128) {
        int cr = idx / P;
        int p  = idx % P;
        int xpos = x0 + p;
        float v = 0.f;
        if (xpos >= 0 && xpos < LIN)
          v = xin[((size_t)b * CIN + ci0 + cr) * LIN + xpos];
        xt[cr][p] = v;
      }
    }
    __syncthreads();

    for (int kk = 0; kk < 3; ++kk) {
      v16bf bh, bl, ah, al;
#pragma unroll
      for (int e = 0; e < 16; ++e) {
        // B fragment: B[k][n] = x[ci0+k][STRIDE*(l0+n)+kk-1]
        int k = e + 16 * half;
        float v = xt[k][STRIDE * nr + kk];
        __bf16 h, l;
        bf_split(v, h, l);
        bh[e] = h; bl[e] = l;
        // A fragment: A[m][k] = W[cobase+m][kk*CIN + ci0 + k]
        int kr = e + 16 * half;
        size_t widx = (size_t)(cobase + nr) * (3 * CIN) + kk * CIN + ci0 + kr;
        ah[e] = w_hi[widx];
        al[e] = w_lo[widx];
      }
      acc = WMMA_BF16(al, bh, acc);  // lo*hi
      acc = WMMA_BF16(ah, bl, acc);  // hi*lo
      acc = WMMA_BF16(ah, bh, acc);  // hi*hi
    }
  }

  // epilogue: D row m = r + 8*half -> cout, col n = nr -> position
#pragma unroll
  for (int r = 0; r < 8; ++r) {
    int co = cobase + r + 8 * half;
    int l  = l0 + nr;
    float v = acc[r] + bias[co];
    if constexpr (MODE == 0) {
      float inv = bn_g[co] / sqrtf(bn_v[co] + 1e-5f);
      v = v * inv + (bn_b[co] - bn_m[co] * inv);
      v = (v > 0.f) ? v : 0.01f * v;
      out[((size_t)b * COUT + co) * LOUT + l] = v;
    } else {
      out[((size_t)b * LOUT + l) * 128 + co] = v;  // z transposed to [n][D]
    }
  }
}

// ---------------------------------------------------------------------------
// VQ: for each of 16 z-rows per block, argmin_m (||e_m||^2 - 2 z.e_m) over
// 4096 codes via bf16x3 WMMA, then gather emb[code] as the quantized output.
// 128 threads = 4 waves, each wave owns a different 16-row code slice.
// ---------------------------------------------------------------------------
__global__ __launch_bounds__(128)
void vq_wmma_kernel(const float* __restrict__ z,        // [N,128]
                    const __bf16* __restrict__ e_hi,
                    const __bf16* __restrict__ e_lo,    // [4096][128]
                    const float* __restrict__ norms,    // [4096]
                    const float* __restrict__ emb,      // [4096][128] f32
                    float* __restrict__ out_codes,      // [N] (as float)
                    float* __restrict__ out_q) {        // [N,128]
  const int tid  = threadIdx.x;
  const int lane = tid & 31;
  const int wv   = tid >> 5;
  const int half = lane >> 4;
  const int nr   = lane & 15;
  const int n0   = blockIdx.x * 16;

  // preload B fragments (z tile) for all 4 K-blocks: B[k][n] = z[n0+n][k]
  v16bf bh[4], bl[4];
#pragma unroll
  for (int kb = 0; kb < 4; ++kb) {
#pragma unroll
    for (int e = 0; e < 16; ++e) {
      int k = kb * 32 + e + 16 * half;
      float v = z[(size_t)(n0 + nr) * 128 + k];
      __bf16 h, l;
      bf_split(v, h, l);
      bh[kb][e] = h; bl[kb][e] = l;
    }
  }

  float bestv = 3.4e38f;
  int   besti = 0;
  for (int m0 = 0; m0 < 4096; m0 += 64) {
    int mrow = m0 + wv * 16 + nr;  // A row for this lane
    v8f acc = {0.f, 0.f, 0.f, 0.f, 0.f, 0.f, 0.f, 0.f};
#pragma unroll
    for (int kb = 0; kb < 4; ++kb) {
      v16bf ah, al;
#pragma unroll
      for (int e = 0; e < 16; ++e) {
        int kr = (e + 16 * half) + kb * 32;
        size_t idx = (size_t)mrow * 128 + kr;
        ah[e] = e_hi[idx];
        al[e] = e_lo[idx];
      }
      acc = WMMA_BF16(al, bh[kb], acc);
      acc = WMMA_BF16(ah, bl[kb], acc);
      acc = WMMA_BF16(ah, bh[kb], acc);
    }
#pragma unroll
    for (int r = 0; r < 8; ++r) {
      int m = m0 + wv * 16 + r + 8 * half;
      float s = norms[m] - 2.f * acc[r];
      if (s < bestv) { bestv = s; besti = m; }  // strict <: keeps smallest m
    }
  }

  __shared__ float sv[128];
  __shared__ int   si[128];
  __shared__ int   codes_s[16];
  sv[tid] = bestv;
  si[tid] = besti;
  __syncthreads();
  if (tid < 16) {
    float bv = 3.4e38f;
    int   bi = 0x7fffffff;
    for (int j = tid; j < 128; j += 16) {  // all threads with column n = tid
      float v = sv[j];
      int   i = si[j];
      if (v < bv || (v == bv && i < bi)) { bv = v; bi = i; }
    }
    codes_s[tid] = bi;
    out_codes[n0 + tid] = (float)bi;
  }
  __syncthreads();
  // gather quantized rows: 128 threads = one D element each, coalesced
  for (int j = 0; j < 16; ++j) {
    int row = codes_s[j];
    out_q[(size_t)(n0 + j) * 128 + tid] = emb[(size_t)row * 128 + tid];
  }
}

// ---------------------------------------------------------------------------
// Host launcher
// ---------------------------------------------------------------------------
extern "C" void kernel_launch(void* const* d_in, const int* in_sizes, int n_in,
                              void* d_out, int out_size, void* d_ws, size_t ws_size,
                              hipStream_t stream) {
  (void)n_in; (void)out_size; (void)ws_size;
  const float* x       = (const float*)d_in[0];
  const float* conv1_w = (const float*)d_in[1];
  const float* conv1_b = (const float*)d_in[2];
  const float* bn1_g   = (const float*)d_in[3];
  const float* bn1_b   = (const float*)d_in[4];
  const float* bn1_m   = (const float*)d_in[5];
  const float* bn1_v   = (const float*)d_in[6];
  const float* conv2_w = (const float*)d_in[7];
  const float* conv2_b = (const float*)d_in[8];
  const float* bn2_g   = (const float*)d_in[9];
  const float* bn2_b   = (const float*)d_in[10];
  const float* bn2_m   = (const float*)d_in[11];
  const float* bn2_v   = (const float*)d_in[12];
  const float* conv3_w = (const float*)d_in[13];
  const float* conv3_b = (const float*)d_in[14];
  const float* emb     = (const float*)d_in[15];

  const int L  = 4096, L1 = 2048, L2 = 1024;
  const int Bn = in_sizes[0] / (256 * L);     // batch (32)
  const int Nz = Bn * L2;                     // flat z rows (32768)

  // workspace layout (bytes)
  char* ws = (char*)d_ws;
  const size_t W1SZ = 256 * 768, W2SZ = 512 * 768, W3SZ = 128 * 1536, ESZ = 4096 * 128;
  __bf16* w1_hi = (__bf16*)(ws);
  __bf16* w1_lo = w1_hi + W1SZ;
  __bf16* w2_hi = w1_lo + W1SZ;
  __bf16* w2_lo = w2_hi + W2SZ;
  __bf16* w3_hi = w2_lo + W2SZ;
  __bf16* w3_lo = w3_hi + W3SZ;
  __bf16* e_hi  = w3_lo + W3SZ;
  __bf16* e_lo  = e_hi + ESZ;
  float*  norms = (float*)(e_lo + ESZ);
  float*  h1    = norms + 4096;                           // [B,256,L1] f32
  float*  h2    = h1 + (size_t)Bn * 256 * L1;             // [B,512,L2] f32
  float*  zbuf  = h1;                                     // alias: h1 dead by conv3

  float* out_codes = (float*)d_out;
  float* out_q     = out_codes + Nz;

  // weight / codebook packing
  pack_w_kernel<<<(256 * 256 * 3 + 255) / 256, 256, 0, stream>>>(conv1_w, w1_hi, w1_lo, 256, 256);
  pack_w_kernel<<<(512 * 256 * 3 + 255) / 256, 256, 0, stream>>>(conv2_w, w2_hi, w2_lo, 512, 256);
  pack_w_kernel<<<(128 * 512 * 3 + 255) / 256, 256, 0, stream>>>(conv3_w, w3_hi, w3_lo, 128, 512);
  pack_emb_kernel<<<4096, 128, 0, stream>>>(emb, e_hi, e_lo, norms);

  // encoder
  conv_wmma_kernel<256, 256, 2, 0><<<dim3(Bn * L1 / 16, 4), 128, 0, stream>>>(
      x, w1_hi, w1_lo, conv1_b, bn1_g, bn1_b, bn1_m, bn1_v, h1, L, L1);
  conv_wmma_kernel<256, 512, 2, 0><<<dim3(Bn * L2 / 16, 8), 128, 0, stream>>>(
      h1, w2_hi, w2_lo, conv2_b, bn2_g, bn2_b, bn2_m, bn2_v, h2, L1, L2);
  conv_wmma_kernel<512, 128, 1, 1><<<dim3(Bn * L2 / 16, 2), 128, 0, stream>>>(
      h2, w3_hi, w3_lo, conv3_b, nullptr, nullptr, nullptr, nullptr, zbuf, L2, L2);

  // vector quantization
  vq_wmma_kernel<<<Nz / 16, 128, 0, stream>>>(zbuf, e_hi, e_lo, norms, emb,
                                              out_codes, out_q);
}
